// SparseCore_27754078667462
// MI455X (gfx1250) — matrix operand order repaired and checked by reading, not compiled, not run on hardware
//
#include <hip/hip_runtime.h>
#include <hip/hip_bf16.h>
#include <math.h>

// ---------------- problem constants ----------------
#define VOCAB   32000
#define DM      1024
#define DFF     4096
#define NLAYERS 2
#define NEXP    8
#define TOPK    2
#define BATCH   4
#define SEQ     2048
#define TTOK    (BATCH * SEQ)          // 8192 tokens
#define TD      ((size_t)TTOK * DM)    // 8,388,608 activation elems
#define NROWS   (TTOK * TOPK)          // 16384 routed rows per layer
#define BM      64                     // GEMM row tile
#define BN      256                    // GEMM col tile
#define BK      32                     // WMMA K step (f16)
#define R_CAP   (NROWS + NEXP * BM)    // 16896 rows incl. per-expert padding
#define ROW_TILES (R_CAP / BM)         // 264
#define LN_EPS  1e-5f

typedef _Float16 v8h  __attribute__((ext_vector_type(8)));
typedef _Float16 v16h __attribute__((ext_vector_type(16)));
typedef float    v8f  __attribute__((ext_vector_type(8)));

static __device__ __forceinline__ v16h ld_frag(const _Float16* p0, const _Float16* p1) {
    v8h lo = *(const v8h*)p0;
    v8h hi = *(const v8h*)p1;
    return __builtin_shufflevector(lo, hi, 0,1,2,3,4,5,6,7,8,9,10,11,12,13,14,15);
}

static __device__ __forceinline__ float f4_elem(const float4 v, int j) {
    // j is a compile-time constant after unrolling
    return (j == 0) ? v.x : (j == 1) ? v.y : (j == 2) ? v.z : v.w;
}

// ---------------- phase 0: embedding gather + symbolic add ----------------
__global__ void k_embed(const int* __restrict__ tokens, const float* __restrict__ sym,
                        const float* __restrict__ emb, float* __restrict__ x) {
    size_t i = (size_t)blockIdx.x * blockDim.x + threadIdx.x;
    if (i >= TD) return;
    int t = (int)(i >> 10);
    int d = (int)(i & 1023);
    int b = t >> 11;                   // t / SEQ
    x[i] = emb[(size_t)tokens[t] * DM + d] + sym[(size_t)b * DM + d];
}

// ---------------- per-layer reset of scratch state ----------------
__global__ void k_reset(float* __restrict__ xnext, int* __restrict__ token_ids,
                        float* __restrict__ row_w, int* __restrict__ cnt, int* __restrict__ cursor) {
    size_t i = (size_t)blockIdx.x * blockDim.x + threadIdx.x;
    if (i < TD) xnext[i] = 0.0f;
    if (i < R_CAP) { token_ids[i] = -1; row_w[i] = 0.0f; }
    if (i < NEXP)  { cnt[i] = 0; cursor[i] = 0; }
}

// ---------------- gating: softmax + top-2, one wave per token ----------------
__global__ void k_gate(const float* __restrict__ x, const float* __restrict__ gw,
                       const float* __restrict__ gb, int* __restrict__ topk_e,
                       float* __restrict__ topk_w, int* __restrict__ cnt) {
    int lane = threadIdx.x & 31;
    int wid  = threadIdx.x >> 5;
    int t = blockIdx.x * 8 + wid;      // 8 waves / block
    if (t >= TTOK) return;
    const float* xr = x + (size_t)t * DM;
    float lg[NEXP];
    #pragma unroll
    for (int e = 0; e < NEXP; ++e) lg[e] = 0.0f;
    for (int d = lane; d < DM; d += 32) {
        float xv = xr[d];
        const float4 g0 = *(const float4*)(gw + (size_t)d * NEXP);
        const float4 g1 = *(const float4*)(gw + (size_t)d * NEXP + 4);
        lg[0] += xv * g0.x; lg[1] += xv * g0.y; lg[2] += xv * g0.z; lg[3] += xv * g0.w;
        lg[4] += xv * g1.x; lg[5] += xv * g1.y; lg[6] += xv * g1.z; lg[7] += xv * g1.w;
    }
    #pragma unroll
    for (int off = 16; off > 0; off >>= 1) {
        #pragma unroll
        for (int e = 0; e < NEXP; ++e) lg[e] += __shfl_xor(lg[e], off, 32);
    }
    if (lane == 0) {
        #pragma unroll
        for (int e = 0; e < NEXP; ++e) lg[e] += gb[e];
        float mx = lg[0];
        #pragma unroll
        for (int e = 1; e < NEXP; ++e) mx = fmaxf(mx, lg[e]);
        float p[NEXP], sum = 0.0f;
        #pragma unroll
        for (int e = 0; e < NEXP; ++e) { p[e] = expf(lg[e] - mx); sum += p[e]; }
        float inv = 1.0f / sum;
        #pragma unroll
        for (int e = 0; e < NEXP; ++e) p[e] *= inv;
        // top-2, strict > so ties pick lower index (matches lax.top_k)
        float m0 = -1.0f, m1 = -1.0f; int i0 = 0, i1 = 0;
        #pragma unroll
        for (int e = 0; e < NEXP; ++e) {
            if (p[e] > m0)      { m1 = m0; i1 = i0; m0 = p[e]; i0 = e; }
            else if (p[e] > m1) { m1 = p[e]; i1 = e; }
        }
        topk_e[2*t]   = i0; topk_w[2*t]   = m0;
        topk_e[2*t+1] = i1; topk_w[2*t+1] = m1;
        atomicAdd(&cnt[i0], 1);
        atomicAdd(&cnt[i1], 1);
    }
}

// ---------------- tile-padded exclusive scan of expert counts ----------------
__global__ void k_offsets(const int* __restrict__ cnt, int* __restrict__ off) {
    if (threadIdx.x == 0 && blockIdx.x == 0) {
        int o = 0;
        off[0] = 0;
        #pragma unroll
        for (int e = 0; e < NEXP; ++e) { o += (cnt[e] + BM - 1) & ~(BM - 1); off[e + 1] = o; }
    }
}

// ---------------- build routed row lists ----------------
__global__ void k_scatter(const int* __restrict__ topk_e, const float* __restrict__ topk_w,
                          const int* __restrict__ off, int* __restrict__ cursor,
                          int* __restrict__ token_ids, float* __restrict__ row_w) {
    int t = blockIdx.x * blockDim.x + threadIdx.x;
    if (t >= TTOK) return;
    #pragma unroll
    for (int k = 0; k < TOPK; ++k) {
        int e = topk_e[2*t + k];
        int pos = atomicAdd(&cursor[e], 1);
        int r = off[e] + pos;
        token_ids[r] = t;
        row_w[r] = topk_w[2*t + k];
    }
}

// ---------------- GEMM1: H = gelu(gather(x) @ W1 + b1), f16 WMMA ----------------
__global__ void __launch_bounds__(256)
k_gemm1(const float* __restrict__ x, const float* __restrict__ w1, const float* __restrict__ b1,
        const int* __restrict__ token_ids, const int* __restrict__ off, _Float16* __restrict__ H) {
    // double-buffered staging tiles (stride 40 halves = 80B -> 16B aligned, conflict-free)
    __shared__ __align__(16) _Float16 As[2][BM][40];   // 64 rows  x 32 K
    __shared__ __align__(16) _Float16 Bs[2][BN][40];   // 256 cols x 32 K (transposed)

    const int row0 = blockIdx.x * BM;
    const int n0   = blockIdx.y * BN;
    int e = 0;
    #pragma unroll
    for (int i = 1; i < NEXP; ++i) if (row0 >= off[i]) e = i;
    const float* W  = w1 + (size_t)e * DM * DFF;
    const float* bb = b1 + (size_t)e * DFF;

    const int tid  = threadIdx.x;
    const int lane = tid & 31;
    const int wave = tid >> 5;
    const int wm = wave & 1, wn = wave >> 1;           // 2x4 wave grid -> [32 x 64] per wave

    // A staging: 8 contiguous floats per thread
    const int arow = tid >> 2;
    const int ak   = (tid & 3) * 8;
    const int tok  = token_ids[row0 + arow];
    const float* xrow = (tok >= 0) ? (x + (size_t)tok * DM) : nullptr;
    // B staging: thread owns K-run of 8 for 4 columns  (k = bk8 + p, cols bn4..bn4+3)
    const int bn4 = (tid & 63) * 4;
    const int bk8 = (tid >> 6) * 8;

    v8f acc[2][4];
    { v8f vz = {};
      #pragma unroll
      for (int m = 0; m < 2; ++m)
          #pragma unroll
          for (int n = 0; n < 4; ++n) acc[m][n] = vz; }

    const int kbA = (lane < 16) ? 0 : 8;
    const int kbB = (lane < 16) ? 0 : 8;
    const int fr  = lane & 15;

    float4 ra0, ra1, rb[8];

    // ---- prologue: load + stage k0 = 0 into buffer 0 ----
    if (tok >= 0) {
        ra0 = *(const float4*)(xrow + ak);
        ra1 = *(const float4*)(xrow + ak + 4);
    } else { ra0 = make_float4(0,0,0,0); ra1 = ra0; }
    #pragma unroll
    for (int p = 0; p < 8; ++p)
        rb[p] = *(const float4*)(W + (size_t)(bk8 + p) * DFF + n0 + bn4);
    {
        v8h ha;
        ha[0]=(_Float16)ra0.x; ha[1]=(_Float16)ra0.y; ha[2]=(_Float16)ra0.z; ha[3]=(_Float16)ra0.w;
        ha[4]=(_Float16)ra1.x; ha[5]=(_Float16)ra1.y; ha[6]=(_Float16)ra1.z; ha[7]=(_Float16)ra1.w;
        *(v8h*)&As[0][arow][ak] = ha;
        #pragma unroll
        for (int j = 0; j < 4; ++j) {
            v8h hb;
            #pragma unroll
            for (int p = 0; p < 8; ++p) hb[p] = (_Float16)f4_elem(rb[p], j);
            *(v8h*)&Bs[0][bn4 + j][bk8] = hb;
        }
    }

    for (int k0 = 0; k0 < DM; k0 += BK) {
        const int buf = (k0 >> 5) & 1;
        const bool more = (k0 + BK) < DM;
        __syncthreads();
        // issue next tile's global loads (latency hidden under the WMMAs below)
        if (more) {
            const int kn = k0 + BK;
            if (tok >= 0) {
                ra0 = *(const float4*)(xrow + kn + ak);
                ra1 = *(const float4*)(xrow + kn + ak + 4);
            }
            #pragma unroll
            for (int p = 0; p < 8; ++p)
                rb[p] = *(const float4*)(W + (size_t)(kn + bk8 + p) * DFF + n0 + bn4);
        }
        // compute from current buffer
        v16h af[2], bf[4];
        #pragma unroll
        for (int m = 0; m < 2; ++m) {
            int r = wm * 32 + m * 16 + fr;
            af[m] = ld_frag(&As[buf][r][kbA], &As[buf][r][kbA + 16]);
        }
        #pragma unroll
        for (int n = 0; n < 4; ++n) {
            int c = wn * 64 + n * 16 + fr;
            bf[n] = ld_frag(&Bs[buf][c][kbB], &Bs[buf][c][kbB + 16]);
        }
        #pragma unroll
        for (int m = 0; m < 2; ++m)
            #pragma unroll
            for (int n = 0; n < 4; ++n)
                acc[m][n] = __builtin_amdgcn_wmma_f32_16x16x32_f16(
                    false, af[m], false, bf[n], (short)0, acc[m][n], false, false);
        // convert + store next tile into the other buffer
        if (more) {
            v8h ha;
            ha[0]=(_Float16)ra0.x; ha[1]=(_Float16)ra0.y; ha[2]=(_Float16)ra0.z; ha[3]=(_Float16)ra0.w;
            ha[4]=(_Float16)ra1.x; ha[5]=(_Float16)ra1.y; ha[6]=(_Float16)ra1.z; ha[7]=(_Float16)ra1.w;
            *(v8h*)&As[buf ^ 1][arow][ak] = ha;
            #pragma unroll
            for (int j = 0; j < 4; ++j) {
                v8h hb;
                #pragma unroll
                for (int p = 0; p < 8; ++p) hb[p] = (_Float16)f4_elem(rb[p], j);
                *(v8h*)&Bs[buf ^ 1][bn4 + j][bk8] = hb;
            }
        }
    }

    // ---- epilogue: bias + exact GELU, store f16 H ----
    const int cn = lane & 15;
    const int mb = (lane >> 4) << 3;   // 0 or 8
    #pragma unroll
    for (int m = 0; m < 2; ++m) {
        int rbase = row0 + wm * 32 + m * 16 + mb;
        #pragma unroll
        for (int ns = 0; ns < 4; ++ns) {
            int col = n0 + wn * 64 + ns * 16 + cn;
            float bias = bb[col];
            #pragma unroll
            for (int v = 0; v < 8; ++v) {
                float h = acc[m][ns][v] + bias;
                float g = 0.5f * h * (1.0f + erff(h * 0.70710678118654752f));
                H[(size_t)(rbase + v) * DFF + col] = (_Float16)g;
            }
        }
    }
}

// ---------------- GEMM2: y = H @ W2 + b2, scatter w*y into xnext ----------------
__global__ void __launch_bounds__(256)
k_gemm2(const _Float16* __restrict__ H, const float* __restrict__ w2, const float* __restrict__ b2,
        const int* __restrict__ token_ids, const float* __restrict__ row_w,
        const int* __restrict__ off, float* __restrict__ xnext) {
    __shared__ __align__(16) _Float16 As[2][BM][40];
    __shared__ __align__(16) _Float16 Bs[2][BN][40];

    const int row0 = blockIdx.x * BM;
    const int n0   = blockIdx.y * BN;
    int e = 0;
    #pragma unroll
    for (int i = 1; i < NEXP; ++i) if (row0 >= off[i]) e = i;
    const float* W  = w2 + (size_t)e * DFF * DM;
    const float* bb = b2 + (size_t)e * DM;

    const int tid  = threadIdx.x;
    const int lane = tid & 31;
    const int wave = tid >> 5;
    const int wm = wave & 1, wn = wave >> 1;

    const int arow = tid >> 2;
    const int ak   = (tid & 3) * 8;
    const _Float16* hrow = H + (size_t)(row0 + arow) * DFF;
    const int bn4 = (tid & 63) * 4;
    const int bk8 = (tid >> 6) * 8;

    v8f acc[2][4];
    { v8f vz = {};
      #pragma unroll
      for (int m = 0; m < 2; ++m)
          #pragma unroll
          for (int n = 0; n < 4; ++n) acc[m][n] = vz; }

    const int kbA = (lane < 16) ? 0 : 8;
    const int kbB = (lane < 16) ? 0 : 8;
    const int fr  = lane & 15;

    v8h rah;
    float4 rb[8];

    // ---- prologue: stage k0 = 0 into buffer 0 ----
    rah = *(const v8h*)(hrow + ak);
    #pragma unroll
    for (int p = 0; p < 8; ++p)
        rb[p] = *(const float4*)(W + (size_t)(bk8 + p) * DM + n0 + bn4);
    *(v8h*)&As[0][arow][ak] = rah;
    #pragma unroll
    for (int j = 0; j < 4; ++j) {
        v8h hb;
        #pragma unroll
        for (int p = 0; p < 8; ++p) hb[p] = (_Float16)f4_elem(rb[p], j);
        *(v8h*)&Bs[0][bn4 + j][bk8] = hb;
    }

    for (int k0 = 0; k0 < DFF; k0 += BK) {
        const int buf = (k0 >> 5) & 1;
        const bool more = (k0 + BK) < DFF;
        __syncthreads();
        if (more) {
            const int kn = k0 + BK;
            rah = *(const v8h*)(hrow + kn + ak);
            #pragma unroll
            for (int p = 0; p < 8; ++p)
                rb[p] = *(const float4*)(W + (size_t)(kn + bk8 + p) * DM + n0 + bn4);
        }
        v16h af[2], bf[4];
        #pragma unroll
        for (int m = 0; m < 2; ++m) {
            int r = wm * 32 + m * 16 + fr;
            af[m] = ld_frag(&As[buf][r][kbA], &As[buf][r][kbA + 16]);
        }
        #pragma unroll
        for (int n = 0; n < 4; ++n) {
            int c = wn * 64 + n * 16 + fr;
            bf[n] = ld_frag(&Bs[buf][c][kbB], &Bs[buf][c][kbB + 16]);
        }
        #pragma unroll
        for (int m = 0; m < 2; ++m)
            #pragma unroll
            for (int n = 0; n < 4; ++n)
                acc[m][n] = __builtin_amdgcn_wmma_f32_16x16x32_f16(
                    false, af[m], false, bf[n], (short)0, acc[m][n], false, false);
        if (more) {
            *(v8h*)&As[buf ^ 1][arow][ak] = rah;
            #pragma unroll
            for (int j = 0; j < 4; ++j) {
                v8h hb;
                #pragma unroll
                for (int p = 0; p < 8; ++p) hb[p] = (_Float16)f4_elem(rb[p], j);
                *(v8h*)&Bs[buf ^ 1][bn4 + j][bk8] = hb;
            }
        }
    }

    // ---- epilogue: bias, scale by routing weight, deterministic 2-way atomic scatter ----
    const int cn = lane & 15;
    const int mb = (lane >> 4) << 3;
    #pragma unroll
    for (int m = 0; m < 2; ++m) {
        int rbase = row0 + wm * 32 + m * 16 + mb;
        int   tks[8];
        float rws[8];
        #pragma unroll
        for (int v = 0; v < 8; ++v) { tks[v] = token_ids[rbase + v]; rws[v] = row_w[rbase + v]; }
        #pragma unroll
        for (int ns = 0; ns < 4; ++ns) {
            int col = n0 + wn * 64 + ns * 16 + cn;
            float bias = bb[col];
            #pragma unroll
            for (int v = 0; v < 8; ++v) {
                if (tks[v] >= 0)
                    atomicAdd(xnext + (size_t)tks[v] * DM + col, (acc[m][ns][v] + bias) * rws[v]);
            }
        }
    }
}

// ---------------- final LayerNorm (two-pass, matches reference) ----------------
__global__ void k_ln(const float* __restrict__ x, const float* __restrict__ g,
                     const float* __restrict__ b, float* __restrict__ out) {
    __shared__ float s1[256];
    const int t = blockIdx.x;
    const int tid = threadIdx.x;
    const float* xr = x + (size_t)t * DM;
    float s = 0.0f;
    #pragma unroll
    for (int i = 0; i < 4; ++i) s += xr[tid + i * 256];
    s1[tid] = s; __syncthreads();
    for (int st = 128; st > 0; st >>= 1) { if (tid < st) s1[tid] += s1[tid + st]; __syncthreads(); }
    const float mu = s1[0] * (1.0f / DM);
    __syncthreads();
    float ss = 0.0f;
    #pragma unroll
    for (int i = 0; i < 4; ++i) { float d = xr[tid + i * 256] - mu; ss += d * d; }
    s1[tid] = ss; __syncthreads();
    for (int st = 128; st > 0; st >>= 1) { if (tid < st) s1[tid] += s1[tid + st]; __syncthreads(); }
    const float r = rsqrtf(s1[0] * (1.0f / DM) + LN_EPS);
    #pragma unroll
    for (int i = 0; i < 4; ++i) {
        int d = tid + i * 256;
        out[(size_t)t * DM + d] = (xr[d] - mu) * r * g[d] + b[d];
    }
}

// ---------------- host launcher ----------------
extern "C" void kernel_launch(void* const* d_in, const int* in_sizes, int n_in,
                              void* d_out, int out_size, void* d_ws, size_t ws_size,
                              hipStream_t stream) {
    const int*   tokens = (const int*)  d_in[0];
    const float* sym    = (const float*)d_in[1];
    const float* emb    = (const float*)d_in[2];
    const float* gate_w = (const float*)d_in[3];
    const float* gate_b = (const float*)d_in[4];
    const float* w1     = (const float*)d_in[5];
    const float* b1     = (const float*)d_in[6];
    const float* w2     = (const float*)d_in[7];
    const float* b2     = (const float*)d_in[8];
    const float* ln_g   = (const float*)d_in[9];
    const float* ln_b   = (const float*)d_in[10];

    char* ws = (char*)d_ws;
    size_t o = 0;
    auto take = [&](size_t bytes) -> void* {
        void* p = ws + o;
        o = (o + bytes + 255) & ~(size_t)255;
        return p;
    };
    float*     xA        = (float*)    take(TD * sizeof(float));
    float*     xB        = (float*)    take(TD * sizeof(float));
    _Float16*  Hbuf      = (_Float16*) take((size_t)R_CAP * DFF * sizeof(_Float16));
    int*       token_ids = (int*)      take(R_CAP * sizeof(int));
    float*     row_w     = (float*)    take(R_CAP * sizeof(float));
    int*       topk_e    = (int*)      take(2 * TTOK * sizeof(int));
    float*     topk_w    = (float*)    take(2 * TTOK * sizeof(float));
    int*       cnt       = (int*)      take(NEXP * sizeof(int));
    int*       cursor    = (int*)      take(NEXP * sizeof(int));
    int*       offs      = (int*)      take((NEXP + 1) * sizeof(int));

    const int elem_blocks = (int)((TD + 255) / 256);

    k_embed<<<elem_blocks, 256, 0, stream>>>(tokens, sym, emb, xA);

    float* cur = xA;
    float* nxt = xB;
    for (int l = 0; l < NLAYERS; ++l) {
        k_reset<<<elem_blocks, 256, 0, stream>>>(nxt, token_ids, row_w, cnt, cursor);
        k_gate<<<TTOK / 8, 256, 0, stream>>>(cur, gate_w + (size_t)l * DM * NEXP,
                                             gate_b + (size_t)l * NEXP, topk_e, topk_w, cnt);
        k_offsets<<<1, 1, 0, stream>>>(cnt, offs);
        k_scatter<<<TTOK / 256, 256, 0, stream>>>(topk_e, topk_w, offs, cursor, token_ids, row_w);
        k_gemm1<<<dim3(ROW_TILES, DFF / BN), 256, 0, stream>>>(
            cur, w1 + (size_t)l * NEXP * DM * DFF, b1 + (size_t)l * NEXP * DFF,
            token_ids, offs, Hbuf);
        k_gemm2<<<dim3(ROW_TILES, DM / BN), 256, 0, stream>>>(
            Hbuf, w2 + (size_t)l * NEXP * DFF * DM, b2 + (size_t)l * NEXP * DM,
            token_ids, row_w, offs, nxt);
        float* tmp = cur; cur = nxt; nxt = tmp;
    }

    k_ln<<<TTOK, 256, 0, stream>>>(cur, ln_g, ln_b, (float*)d_out);
}
